// SwinTransformerBlock_78829829751137
// MI455X (gfx1250) — matrix operand order repaired and checked by reading, not compiled, not run on hardware
//
#include <hip/hip_runtime.h>
#include <hip/hip_bf16.h>
#include <cmath>

// ---------------- types ----------------
typedef _Float16 half_t;
typedef __attribute__((ext_vector_type(16))) _Float16 v16h;
typedef __attribute__((ext_vector_type(8)))  _Float16 v8h;
typedef __attribute__((ext_vector_type(8)))  float    v8f;
typedef int v4i_vs __attribute__((vector_size(16)));   // matches builtin param

// ---------------- CDNA5 async global->LDS copy (guarded) -------------------
#if defined(__has_builtin)
#if __has_builtin(__builtin_amdgcn_global_load_async_to_lds_b128) && \
    __has_builtin(__builtin_amdgcn_s_wait_asynccnt)
#define USE_ASYNC_LDS 1
#endif
#endif
#ifndef USE_ASYNC_LDS
#define USE_ASYNC_LDS 0
#endif

#if USE_ASYNC_LDS
#define ASYNC_CP16(gsrc, ldst)                                               \
    __builtin_amdgcn_global_load_async_to_lds_b128(                          \
        (__attribute__((address_space(1))) v4i_vs*)(gsrc),                   \
        (__attribute__((address_space(3))) v4i_vs*)(ldst), 0, 0)
#define ASYNC_WAIT() __builtin_amdgcn_s_wait_asynccnt(0)
#else
#define ASYNC_CP16(gsrc, ldst) (*(v8h*)(ldst) = *(const v8h*)(gsrc))
#define ASYNC_WAIT() ((void)0)
#endif

// ---------------- problem constants ----------------
#define BB      32
#define HH      56
#define WW      56
#define CC      192
#define WS      7
#define SS      3
#define NHEAD   6
#define HD      32          // CC / NHEAD
#define NTOK    49          // WS*WS
#define NPAD    64          // padded window tokens
#define NWIN    64          // (H/WS)*(W/WS)
#define TOKENS  (BB*HH*WW)  // 100352
#define C3      (3*CC)      // 576
#define HIDDEN  (4*CC)      // 768
#define QSCALE  0.17677669529663689f  // 1/sqrt(32)

// ---------------- fragment loaders (documented CDNA5 WMMA layouts) ---------
// A 16x32 f16: lane L: row = L&15, kbase = (L>>4)*8; halves [kbase..kbase+7]
// then [kbase+16..kbase+23].  Caller passes &row[kbase].
static __device__ __forceinline__ v16h ld_afrag(const half_t* p) {
    v8h lo = *(const v8h*)(p);
    v8h hi = *(const v8h*)(p + 16);
    return __builtin_shufflevector(lo, hi, 0,1,2,3,4,5,6,7,8,9,10,11,12,13,14,15);
}
// B 32x16 f16 staged K-contiguous per column: lane L: kbase = (L>>4)*8;
// halves [kbase..kbase+7] then [kbase+16..kbase+23].  Caller passes &col[kbase].
static __device__ __forceinline__ v16h ld_bfrag(const half_t* p) {
    v8h lo = *(const v8h*)(p);
    v8h hi = *(const v8h*)(p + 16);
    return __builtin_shufflevector(lo, hi, 0,1,2,3,4,5,6,7,8,9,10,11,12,13,14,15);
}
static __device__ __forceinline__ v8f wmma_f16(v16h a, v16h b, v8f c) {
    return __builtin_amdgcn_wmma_f32_16x16x32_f16(false, a, false, b, (short)0, c, false, false);
}

// ---------------- LayerNorm (+optional shift/window-partition gather) ------
// grid = TOKENS blocks, 32 threads (one wave per token).
__global__ __launch_bounds__(32)
void ln_kernel(const float* __restrict__ x, const float* __restrict__ g,
               const float* __restrict__ b, half_t* __restrict__ out, int do_shift)
{
    const int r    = blockIdx.x;      // output row
    const int lane = threadIdx.x;
    size_t src;
    if (do_shift) {
        int win = r / NTOK, t = r % NTOK;
        int bi  = win / NWIN, wi = win % NWIN;
        int wh  = wi >> 3, ww = wi & 7;
        int sh  = wh * WS + t / WS;
        int sw  = ww * WS + t % WS;
        int h   = (sh + SS) % HH;
        int w   = (sw + SS) % WW;
        src = (size_t)bi * (HH * WW) + (size_t)h * WW + w;
    } else {
        src = (size_t)r;
    }
    const float* xp = x + src * CC;
    float v[6];
    float s = 0.f;
#pragma unroll
    for (int j = 0; j < 6; ++j) { v[j] = xp[lane + j * 32]; s += v[j]; }
#pragma unroll
    for (int m = 16; m >= 1; m >>= 1) s += __shfl_xor(s, m, 32);
    const float mean = s * (1.0f / CC);
    float vs = 0.f;
#pragma unroll
    for (int j = 0; j < 6; ++j) { float d = v[j] - mean; vs += d * d; }
#pragma unroll
    for (int m = 16; m >= 1; m >>= 1) vs += __shfl_xor(vs, m, 32);
    const float rstd = rsqrtf(vs * (1.0f / CC) + 1e-5f);
    half_t* op = out + (size_t)r * CC;
#pragma unroll
    for (int j = 0; j < 6; ++j) {
        int c = lane + j * 32;
        op[c] = (half_t)((v[j] - mean) * rstd * g[c] + b[c]);
    }
}

// ---------------- f32 -> f16 weight convert --------------------------------
__global__ __launch_bounds__(256)
void cvt_kernel(const float* __restrict__ in, half_t* __restrict__ out, int n)
{
    int i = blockIdx.x * 256 + threadIdx.x;
    if (i < n) out[i] = (half_t)in[i];
}

// ---------------- bias + shift-mask precompute -----------------------------
// bm[wi][head][i][j]; -1e30 padding outside the 49x49 valid region.
__device__ __forceinline__ int region_id(int p) { return p < 49 ? 0 : (p < 53 ? 1 : 2); }

__global__ __launch_bounds__(128)
void bm_kernel(const float* __restrict__ rel_table, float* __restrict__ bm)
{
    const int wi   = blockIdx.x;   // window-in-image 0..63
    const int head = blockIdx.y;   // 0..5
    const int wh = wi >> 3, wwp = wi & 7;
    float* out = bm + ((size_t)wi * NHEAD + head) * NPAD * NPAD;
    for (int idx = threadIdx.x; idx < NPAD * NPAD; idx += 128) {
        int i = idx >> 6, j = idx & 63;
        float val;
        if (i < NTOK && j < NTOK) {
            // relative position bias (meshgrid-'xy' faithful, clipped gather)
            int ridx = 13 * ((i % 7 - j % 7 + 6) + (i / 7 - j / 7 + 6));
            ridx = ridx < 0 ? 0 : (ridx > 168 ? 168 : ridx);
            float bias = rel_table[ridx * NHEAD + head];
            int li = region_id(wh * WS + i / 7) * 3 + region_id(wwp * WS + i % 7);
            int lj = region_id(wh * WS + j / 7) * 3 + region_id(wwp * WS + j % 7);
            val = bias + (li != lj ? -100.0f : 0.0f);
        } else {
            val = -1e30f;
        }
        out[idx] = val;
    }
}

// ---------------- generic WMMA GEMM --------------------------------------
// 128x64 block tile, BK=32, double-buffered LDS.
// A-tile staged via async global->LDS (when available); B-tile register-
// transposed.  8 waves: wave w computes rows w*16..w*16+15 over 64 columns
// (4 WMMAs per K-step, single A-fragment reused across 4 B-fragments).
enum { EPI_QKV = 0, EPI_GELU = 1, EPI_PROJ = 2, EPI_FC2 = 3 };

template <int EPI>
__global__ __launch_bounds__(256)
void gemm16(const half_t* __restrict__ A, const half_t* __restrict__ Wt,
            const float* __restrict__ bias, half_t* __restrict__ out16,
            float* __restrict__ out32, const float* __restrict__ resid,
            int N, int K)
{
    __shared__ half_t As[2][128][32];   // [row][k]  row-major
    __shared__ half_t Bs[2][64][32];    // [col][k]  K-contiguous per column

    const int tid  = threadIdx.x;
    const int lane = tid & 31;
    const int w    = tid >> 5;                 // M-tile 0..7
    const int m0   = blockIdx.y * 128;
    const int n0   = blockIdx.x * 64;

    // staging indices
    const int arow = tid >> 1;                 // 0..127
    const int aks  = (tid & 1) * 16;           // 0 / 16
    const int bkk  = tid >> 3;                 // 0..31
    const int bcs  = (tid & 7) * 8;            // 0..56

    const half_t* aptr = A + (size_t)(m0 + arow) * K + aks;
    const half_t* bptr = Wt + (size_t)bkk * N + n0 + bcs;

    v8h rb;
    // A tile: copy straight into LDS (async path writes LDS directly).
    auto issue_A = [&](int k0, int buf) {
        ASYNC_CP16(aptr + k0,     &As[buf][arow][aks]);
        ASYNC_CP16(aptr + k0 + 8, &As[buf][arow][aks + 8]);
    };
    auto load_B = [&](int k0) { rb = *(const v8h*)(bptr + (size_t)k0 * N); };
    auto store_B = [&](int buf) {
#pragma unroll
        for (int j = 0; j < 8; ++j) Bs[buf][bcs + j][bkk] = rb[j];
    };

    v8f acc[4] = {};
    issue_A(0, 0);
    load_B(0);
    store_B(0);
    ASYNC_WAIT();
    __syncthreads();

    const int kb  = (lane >> 4) * 8;
    const int kb2 = (lane >> 4) * 8;
    int buf = 0;
    for (int k0 = 0; k0 < K; k0 += 32) {
        const bool last = (k0 + 32 >= K);
        if (!last) {
            issue_A(k0 + 32, buf ^ 1);                   // overlap with WMMAs
            load_B(k0 + 32);
            __builtin_prefetch(aptr + k0 + 64, 0, 3);    // hint next-next tile
            __builtin_prefetch(bptr + (size_t)(k0 + 64) * N, 0, 3);
        }
        v16h afrag = ld_afrag(&As[buf][w * 16 + (lane & 15)][kb]);
#pragma unroll
        for (int nt = 0; nt < 4; ++nt) {
            v16h b = ld_bfrag(&Bs[buf][nt * 16 + (lane & 15)][kb2]);
            acc[nt] = wmma_f16(afrag, b, acc[nt]);
        }
        if (!last) {
            store_B(buf ^ 1);
            ASYNC_WAIT();
            __syncthreads();
            buf ^= 1;
        }
    }

    // ------------- epilogue -------------
#pragma unroll
    for (int nt = 0; nt < 4; ++nt) {
        const int gn = n0 + nt * 16 + (lane & 15);
        const float bv = bias[gn];
#pragma unroll
        for (int r = 0; r < 8; ++r) {
            const int gm = m0 + w * 16 + r + 8 * (lane >> 4);
            float val = acc[nt][r] + bv;
            if constexpr (EPI == EPI_QKV) {
                if (gn < CC) val *= QSCALE;                 // pre-scale q
                out16[(size_t)gm * N + gn] = (half_t)val;
            } else if constexpr (EPI == EPI_GELU) {
                val = 0.5f * val * (1.0f + erff(val * 0.7071067811865476f));
                out16[(size_t)gm * N + gn] = (half_t)val;
            } else if constexpr (EPI == EPI_PROJ) {
                // window-reverse + roll(+SS) scatter + residual
                int win = gm / NTOK, t = gm % NTOK;
                int bi  = win / NWIN, wi = win % NWIN;
                int wh  = wi >> 3, ww = wi & 7;
                int sh  = wh * WS + t / WS;
                int sw  = ww * WS + t % WS;
                int h   = (sh + SS) % HH;
                int wc  = (sw + SS) % WW;
                size_t dst = ((size_t)bi * (HH * WW) + (size_t)h * WW + wc) * CC + gn;
                out32[dst] = resid[dst] + val;
            } else {  // EPI_FC2
                size_t dst = (size_t)gm * N + gn;
                out32[dst] = resid[dst] + val;
            }
        }
    }
}

// ---------------- window attention --------------------------------------
// grid = (2048 windows, 6 heads), 128 threads (4 waves).
__global__ __launch_bounds__(128)
void attn_kernel(const half_t* __restrict__ qkv, const float* __restrict__ bm,
                 half_t* __restrict__ o)
{
    __shared__ half_t qs[NPAD][HD];
    __shared__ half_t ks[NPAD][HD];
    __shared__ half_t vt[HD][NPAD];        // v transposed: vt[d][tok]
    __shared__ float  sc[NPAD][NPAD];
    __shared__ half_t at[NPAD][NPAD];
    __shared__ float  red[NPAD][2];

    const int win  = blockIdx.x;
    const int head = blockIdx.y;
    const int tid  = threadIdx.x;
    const int lane = tid & 31;
    const int wv   = tid >> 5;             // wave id = M-tile

    // ---- stage q/k async; v transposed synchronously; zero-pad 49..63 ----
    {
        const int tok = tid >> 1;
        const int dh  = (tid & 1) * 16;
        const size_t base = (size_t)win * NTOK * C3 + (size_t)head * HD + dh;
        v8h z = {};
        if (tok < NTOK) {
            const half_t* p = qkv + base + (size_t)tok * C3;
            ASYNC_CP16(p + 0 * CC,     &qs[tok][dh]);
            ASYNC_CP16(p + 0 * CC + 8, &qs[tok][dh + 8]);
            ASYNC_CP16(p + 1 * CC,     &ks[tok][dh]);
            ASYNC_CP16(p + 1 * CC + 8, &ks[tok][dh + 8]);
            v8h v0 = *(const v8h*)(p + 2 * CC);
            v8h v1 = *(const v8h*)(p + 2 * CC + 8);
#pragma unroll
            for (int jj = 0; jj < 8; ++jj) {
                vt[dh + jj][tok]     = v0[jj];
                vt[dh + 8 + jj][tok] = v1[jj];
            }
        } else {
            *(v8h*)&qs[tok][dh] = z; *(v8h*)&qs[tok][dh + 8] = z;
            *(v8h*)&ks[tok][dh] = z; *(v8h*)&ks[tok][dh + 8] = z;
#pragma unroll
            for (int jj = 0; jj < 16; ++jj) vt[dh + jj][tok] = (half_t)0.f;
        }
    }
    ASYNC_WAIT();
    __syncthreads();

    const float* bmp = bm + ((size_t)(win & (NWIN - 1)) * NHEAD + head) * NPAD * NPAD;

    // ---- scores = q @ k^T  (16 rows per wave, K=32) + bias/mask table ----
    {
        const int arow = wv * 16 + (lane & 15);
        const int kb   = (lane >> 4) * 8;
        const int kb2  = (lane >> 4) * 8;
        v16h a = ld_afrag(&qs[arow][kb]);
#pragma unroll
        for (int nt = 0; nt < 4; ++nt) {
            const int colK = nt * 16 + (lane & 15);
            v16h b = ld_bfrag(&ks[colK][kb2]);      // B[kk][n] = k[n][kk]
            v8f z = {};
            v8f acc = wmma_f16(a, b, z);
#pragma unroll
            for (int r = 0; r < 8; ++r) {
                const int i = wv * 16 + r + 8 * (lane >> 4);
                const int j = nt * 16 + (lane & 15);
                sc[i][j] = acc[r] + bmp[i * NPAD + j];   // L2-resident table
            }
        }
    }
    __syncthreads();

    // ---- softmax: 2 threads per row, 32 columns each ----
    {
        const int row = tid >> 1;
        const int hf  = tid & 1;
        const int c0  = hf * 32;
        float pmx = -1e30f;
#pragma unroll 8
        for (int j = 0; j < 32; ++j) pmx = fmaxf(pmx, sc[row][c0 + j]);
        red[row][hf] = pmx;
        __syncthreads();
        const float mx = fmaxf(red[row][0], red[row][1]);
        __syncthreads();
        float psum = 0.f;
#pragma unroll 8
        for (int j = 0; j < 32; ++j) {
            float e = expf(sc[row][c0 + j] - mx);
            sc[row][c0 + j] = e;
            psum += e;
        }
        red[row][hf] = psum;
        __syncthreads();
        const float inv = 1.0f / (red[row][0] + red[row][1]);
#pragma unroll 8
        for (int j = 0; j < 32; ++j) at[row][c0 + j] = (half_t)(sc[row][c0 + j] * inv);
    }
    __syncthreads();

    // ---- out = attn @ v  (K=64, two WMMA K-steps; 2 N-tiles of 16) ----
    {
        const int arow = wv * 16 + (lane & 15);
        const int kb   = (lane >> 4) * 8;
        const int kb2  = (lane >> 4) * 8;
        const int colv = lane & 15;
#pragma unroll
        for (int nt = 0; nt < 2; ++nt) {
            v8f acc = {};
#pragma unroll
            for (int s = 0; s < 2; ++s) {
                v16h a = ld_afrag(&at[arow][s * 32 + kb]);
                v16h b = ld_bfrag(&vt[nt * 16 + colv][s * 32 + kb2]);  // contiguous
                acc = wmma_f16(a, b, acc);
            }
#pragma unroll
            for (int r = 0; r < 8; ++r) {
                const int i = wv * 16 + r + 8 * (lane >> 4);
                if (i < NTOK) {
                    o[((size_t)win * NTOK + i) * CC + head * HD + nt * 16 + colv] =
                        (half_t)acc[r];
                }
            }
        }
    }
}

// ---------------- host-side launch ----------------------------------------
extern "C" void kernel_launch(void* const* d_in, const int* in_sizes, int n_in,
                              void* d_out, int out_size, void* d_ws, size_t ws_size,
                              hipStream_t stream)
{
    const float* x        = (const float*)d_in[0];
    const float* norm1_g  = (const float*)d_in[1];
    const float* norm1_b  = (const float*)d_in[2];
    const float* qkv_w    = (const float*)d_in[3];
    const float* qkv_b    = (const float*)d_in[4];
    const float* rel_tab  = (const float*)d_in[5];
    const float* proj_w   = (const float*)d_in[6];
    const float* proj_b   = (const float*)d_in[7];
    const float* norm2_g  = (const float*)d_in[8];
    const float* norm2_b  = (const float*)d_in[9];
    const float* fc1_w    = (const float*)d_in[10];
    const float* fc1_b    = (const float*)d_in[11];
    const float* fc2_w    = (const float*)d_in[12];
    const float* fc2_b    = (const float*)d_in[13];
    float* out            = (float*)d_out;

    // workspace carve
    size_t off = 0;
    auto carve = [&](size_t bytes) -> void* {
        void* p = (char*)d_ws + off;
        off += (bytes + 255) & ~(size_t)255;
        return p;
    };
    half_t* xw      = (half_t*)carve((size_t)TOKENS * CC * 2);
    half_t* qkv16   = (half_t*)carve((size_t)TOKENS * C3 * 2);
    half_t* o16     = (half_t*)carve((size_t)TOKENS * CC * 2);
    float*  xnew    = (float*) carve((size_t)TOKENS * CC * 4);
    half_t* m16     = (half_t*)carve((size_t)TOKENS * CC * 2);
    half_t* h116    = (half_t*)carve((size_t)TOKENS * HIDDEN * 2);
    half_t* qkvw16  = (half_t*)carve((size_t)CC * C3 * 2);
    half_t* projw16 = (half_t*)carve((size_t)CC * CC * 2);
    half_t* fc1w16  = (half_t*)carve((size_t)CC * HIDDEN * 2);
    half_t* fc2w16  = (half_t*)carve((size_t)HIDDEN * CC * 2);
    float*  bmbuf   = (float*) carve((size_t)NWIN * NHEAD * NPAD * NPAD * 4);
    (void)ws_size; (void)in_sizes; (void)n_in; (void)out_size;

    // weight conversion to f16
    auto cvt = [&](const float* src, half_t* dst, int n) {
        cvt_kernel<<<(n + 255) / 256, 256, 0, stream>>>(src, dst, n);
    };
    cvt(qkv_w, qkvw16, CC * C3);
    cvt(proj_w, projw16, CC * CC);
    cvt(fc1_w, fc1w16, CC * HIDDEN);
    cvt(fc2_w, fc2w16, HIDDEN * CC);

    // bias + mask table (L2-resident, reused by all 12288 attention blocks)
    bm_kernel<<<dim3(NWIN, NHEAD), 128, 0, stream>>>(rel_tab, bmbuf);

    // 1) LN1 + shift + window partition -> xw (f16)
    ln_kernel<<<TOKENS, 32, 0, stream>>>(x, norm1_g, norm1_b, xw, 1);

    // 2) QKV GEMM (q pre-scaled)
    gemm16<EPI_QKV><<<dim3(C3 / 64, TOKENS / 128), 256, 0, stream>>>(
        xw, qkvw16, qkv_b, qkv16, nullptr, nullptr, C3, CC);

    // 3) window attention
    attn_kernel<<<dim3(BB * NWIN, NHEAD), 128, 0, stream>>>(qkv16, bmbuf, o16);

    // 4) proj GEMM + window reverse + unshift + residual -> xnew (f32)
    gemm16<EPI_PROJ><<<dim3(CC / 64, TOKENS / 128), 256, 0, stream>>>(
        o16, projw16, proj_b, nullptr, xnew, x, CC, CC);

    // 5) LN2 -> m16
    ln_kernel<<<TOKENS, 32, 0, stream>>>(xnew, norm2_g, norm2_b, m16, 0);

    // 6) fc1 GEMM + exact GELU -> h116
    gemm16<EPI_GELU><<<dim3(HIDDEN / 64, TOKENS / 128), 256, 0, stream>>>(
        m16, fc1w16, fc1_b, h116, nullptr, nullptr, HIDDEN, CC);

    // 7) fc2 GEMM + residual -> out
    gemm16<EPI_FC2><<<dim3(CC / 64, TOKENS / 128), 256, 0, stream>>>(
        h116, fc2w16, fc2_b, nullptr, out, xnew, CC, HIDDEN);
}
